// SelfAttentionEncoderBlock_28252294873695
// MI455X (gfx1250) — matrix operand order repaired and checked by reading, not compiled, not run on hardware
//
#include <hip/hip_runtime.h>

#define DI __device__ __forceinline__

typedef __bf16 bf16_t;
typedef bf16_t v16bf __attribute__((ext_vector_type(16)));
typedef bf16_t v8bf  __attribute__((ext_vector_type(8)));
typedef float  v8f   __attribute__((ext_vector_type(8)));

static constexpr int  Bb   = 8;
static constexpr int  Nn   = 1024;   // NH*NW
static constexpr int  Cc   = 384;
static constexpr int  Hh   = 12;
static constexpr int  HDc  = 32;
static constexpr int  HIDc = 768;
static constexpr float SCALEf = 0.17677669529663687f; // 32^-0.5
static constexpr float EPSf   = 1e-5f;

#define WMMA_BF16(a, b, c) \
  __builtin_amdgcn_wmma_f32_16x16x32_bf16(false, (a), false, (b), (short)0, (c), false, false)

// ---------- scalar helpers ----------
DI bf16_t f2bf(float f) {
  unsigned u = __builtin_bit_cast(unsigned, f);
  unsigned short r = (unsigned short)((u + 0x7FFFu + ((u >> 16) & 1u)) >> 16); // RNE
  return __builtin_bit_cast(bf16_t, r);
}
DI float bf2f(bf16_t b) {
  unsigned u = ((unsigned)__builtin_bit_cast(unsigned short, b)) << 16;
  return __builtin_bit_cast(float, u);
}
DI float gelu_exact(float x) { return 0.5f * x * (1.0f + erff(x * 0.70710678118654752f)); }

// ---------- WMMA fragment loaders (contiguous k order per lane half, not the ISA table order, wave32) ----------
// A (16x32 bf16): lane L -> row = m0+(L&15); k = base .. base+15 contiguous, base=(L>>4)*16
DI v16bf load_a_frag(const bf16_t* A, int m0, int ldk, int k, int lane) {
  int row = m0 + (lane & 15);
  const bf16_t* p = A + (size_t)row * ldk + k + ((lane >> 4) << 4);
  v8bf lo = *reinterpret_cast<const v8bf*>(p);
  v8bf hi = *reinterpret_cast<const v8bf*>(p + 8);
  return __builtin_shufflevector(lo, hi, 0,1,2,3,4,5,6,7,8,9,10,11,12,13,14,15);
}
// B (32x16 bf16) from row-major W[N][K] (B[k][n] = W[n][k]): lane L -> col n0+(L&15), k = kbase + (L>>4)*16 .. +15 contiguous
DI v16bf load_b_frag(const bf16_t* W, int n0, int ldk, int k, int lane) {
  const bf16_t* p = W + (size_t)(n0 + (lane & 15)) * ldk + k + ((lane >> 4) << 4);
  return *reinterpret_cast<const v16bf*>(p);
}

// ---------- 32x64-per-wave GEMM core: C += A[M,K] * W[N,K]^T ----------
// 8 WMMAs per K-step fed by 6 fragment loads -> ~22 FLOP/byte from L1/L2.
DI void gemm_core(const bf16_t* __restrict__ A, const bf16_t* __restrict__ W, int K,
                  int m0, int n0, int lane, v8f (&c)[2][4]) {
#pragma unroll 2
  for (int k = 0; k < K; k += 32) {
    v16bf a0 = load_a_frag(A, m0,      K, k, lane);
    v16bf a1 = load_a_frag(A, m0 + 16, K, k, lane);
#pragma unroll
    for (int j = 0; j < 4; ++j) {
      v16bf bj = load_b_frag(W, n0 + 16 * j, K, k, lane);
      c[0][j] = WMMA_BF16(a0, bj, c[0][j]);
      c[1][j] = WMMA_BF16(a1, bj, c[1][j]);
    }
  }
}

// ---------- small elementwise kernels ----------
__global__ __launch_bounds__(256) void k_cvt_bf16(const float* __restrict__ in,
                                                  bf16_t* __restrict__ out, int n) {
  int i = blockIdx.x * 256 + threadIdx.x;
  if (i < n) out[i] = f2bf(in[i]);
}

// bias[h,q,k] = rel_table[rel_index[q,k], h] + global_bias[q,k]
__global__ __launch_bounds__(256) void k_bias_combine(const int* __restrict__ ri,
                                                      const float* __restrict__ rt,
                                                      const float* __restrict__ gb,
                                                      float* __restrict__ out) {
  int idx = blockIdx.x * 256 + threadIdx.x;
  if (idx >= Nn * Nn) return;
  int r = ri[idx];
  float g = gb[idx];
#pragma unroll
  for (int h = 0; h < Hh; ++h)
    out[(size_t)h * Nn * Nn + idx] = rt[r * Hh + h] + g;
}

// LayerNorm over C=384, one wave per row, bf16 output
__global__ __launch_bounds__(256) void k_ln_bf16(const float* __restrict__ x,
                                                 const float* __restrict__ g,
                                                 const float* __restrict__ b,
                                                 bf16_t* __restrict__ out) {
  int lane = threadIdx.x & 31, wave = threadIdx.x >> 5;
  int row = blockIdx.x * 8 + wave;
  const float* xr = x + (size_t)row * Cc;
  float v[12], s = 0.f, sq = 0.f;
#pragma unroll
  for (int t = 0; t < 12; ++t) {
    v[t] = xr[lane + t * 32];
    s += v[t]; sq += v[t] * v[t];
  }
#pragma unroll
  for (int m = 16; m > 0; m >>= 1) { s += __shfl_xor(s, m, 32); sq += __shfl_xor(sq, m, 32); }
  float mean = s * (1.f / 384.f);
  float var  = sq * (1.f / 384.f) - mean * mean;
  float rstd = rsqrtf(var + EPSf);
#pragma unroll
  for (int t = 0; t < 12; ++t) {
    int i = lane + t * 32;
    out[(size_t)row * Cc + i] = f2bf((v[t] - mean) * rstd * g[i] + b[i]);
  }
}

// depthwise 3x3, dilation 2, pad 2 on 32x32 maps, channels-last, + exact GELU
__global__ __launch_bounds__(256) void k_dwconv_gelu(const bf16_t* __restrict__ in,
                                                     const float* __restrict__ w,
                                                     const float* __restrict__ bias,
                                                     bf16_t* __restrict__ out) {
  int idx = blockIdx.x * 256 + threadIdx.x;
  if (idx >= Bb * Nn * HIDc) return;
  int ch = idx % HIDc;
  int sp = (idx / HIDc) & (Nn - 1);
  int bb = idx / (HIDc * Nn);
  int hh = sp >> 5, ww = sp & 31;
  const float* wc = w + ch * 9;
  float acc = bias[ch];
#pragma unroll
  for (int dy = 0; dy < 3; ++dy) {
    int ih = hh + 2 * (dy - 1);
    if ((unsigned)ih >= 32u) continue;
#pragma unroll
    for (int dx = 0; dx < 3; ++dx) {
      int iw = ww + 2 * (dx - 1);
      if ((unsigned)iw >= 32u) continue;
      acc += wc[dy * 3 + dx] * bf2f(in[((size_t)bb * Nn + ih * 32 + iw) * HIDc + ch]);
    }
  }
  out[idx] = f2bf(gelu_exact(acc));
}

// ---------- GEMM kernels (256 thr = 8 waves arranged 4(M) x 2(N); block tile 128x128) ----------
__global__ __launch_bounds__(256) void k_gemm_gelu_bf16(const bf16_t* __restrict__ A,
                                                        const bf16_t* __restrict__ W,
                                                        const float* __restrict__ bias,
                                                        bf16_t* __restrict__ out,
                                                        int K, int Nc) {
  int lane = threadIdx.x & 31, wave = threadIdx.x >> 5;
  int m0 = blockIdx.y * 128 + (wave & 3) * 32;
  int n0 = blockIdx.x * 128 + (wave >> 2) * 64;
  int half = lane >> 4, l15 = lane & 15;
  v8f c[2][4] = {};
  gemm_core(A, W, K, m0, n0, lane, c);
#pragma unroll
  for (int i = 0; i < 2; ++i)
#pragma unroll
    for (int jn = 0; jn < 4; ++jn) {
      int col = n0 + 16 * jn + l15;
      float bv = bias[col];
#pragma unroll
      for (int j = 0; j < 8; ++j) {
        int row = m0 + 16 * i + half * 8 + j;
        out[(size_t)row * Nc + col] = f2bf(gelu_exact(c[i][jn][j] + bv));
      }
    }
}

__global__ __launch_bounds__(256) void k_gemm_bias_res_f32(const bf16_t* __restrict__ A,
                                                           const bf16_t* __restrict__ W,
                                                           const float* __restrict__ bias,
                                                           const float* __restrict__ res,
                                                           float* __restrict__ out,
                                                           int K, int Nc) {
  int lane = threadIdx.x & 31, wave = threadIdx.x >> 5;
  int m0 = blockIdx.y * 128 + (wave & 3) * 32;
  int n0 = blockIdx.x * 128 + (wave >> 2) * 64;
  int half = lane >> 4, l15 = lane & 15;
  v8f c[2][4] = {};
  gemm_core(A, W, K, m0, n0, lane, c);
#pragma unroll
  for (int i = 0; i < 2; ++i)
#pragma unroll
    for (int jn = 0; jn < 4; ++jn) {
      int col = n0 + 16 * jn + l15;
      float bv = bias[col];
#pragma unroll
      for (int j = 0; j < 8; ++j) {
        size_t o = (size_t)(m0 + 16 * i + half * 8 + j) * Nc + col;
        out[o] = c[i][jn][j] + bv + res[o];
      }
    }
}

// QKV GEMM (Nc=1152). Epilogue: Q scaled by HD^-0.5 -> [b,h,n,d]; K -> [b,h,n,d]; V transposed -> [b,h,d,n]
__global__ __launch_bounds__(256) void k_gemm_qkv(const bf16_t* __restrict__ A,
                                                  const bf16_t* __restrict__ W,
                                                  bf16_t* __restrict__ qb,
                                                  bf16_t* __restrict__ kb,
                                                  bf16_t* __restrict__ vtb) {
  const int K = Cc;
  int lane = threadIdx.x & 31, wave = threadIdx.x >> 5;
  int m0 = blockIdx.y * 128 + (wave & 3) * 32;
  int n0 = blockIdx.x * 128 + (wave >> 2) * 64;
  int half = lane >> 4, l15 = lane & 15;
  v8f c[2][4] = {};
  gemm_core(A, W, K, m0, n0, lane, c);
#pragma unroll
  for (int i = 0; i < 2; ++i)
#pragma unroll
    for (int jn = 0; jn < 4; ++jn) {
      int col = n0 + 16 * jn + l15;
      int sel = col / Cc, rem = col % Cc;
      int h = rem >> 5, d = rem & 31;
#pragma unroll
      for (int j = 0; j < 8; ++j) {
        int row = m0 + 16 * i + half * 8 + j;
        int bidx = row >> 10, qi = row & 1023;
        size_t bh = (size_t)bidx * Hh + h;
        float v = c[i][jn][j];
        if (sel == 0)      qb[(bh * Nn + qi) * HDc + d] = f2bf(v * SCALEf);
        else if (sel == 1) kb[(bh * Nn + qi) * HDc + d] = f2bf(v);
        else               vtb[(bh * HDc + d) * Nn + qi] = f2bf(v);
      }
    }
}

// ---------- flash attention: 128 thr = 4 waves, each wave owns 16 queries ----------
__global__ __launch_bounds__(128) void k_flash_attn(const bf16_t* __restrict__ qbuf,
                                                    const bf16_t* __restrict__ kbuf,
                                                    const bf16_t* __restrict__ vtbuf,
                                                    const float* __restrict__ biasbuf,
                                                    bf16_t* __restrict__ outb) {
  __shared__ bf16_t pshm[4][16 * 32];   // per-wave P staging (C-frag -> A-frag relayout)
  int lane = threadIdx.x & 31, wave = threadIdx.x >> 5;
  int half = lane >> 4, l15 = lane & 15;
  int b = blockIdx.z, h = blockIdx.y;
  size_t bh = (size_t)b * Hh + h;
  int q0 = blockIdx.x * 64 + wave * 16;

  const bf16_t* Q  = qbuf  + bh * Nn * HDc;
  const bf16_t* Kp = kbuf  + bh * Nn * HDc;
  const bf16_t* Vt = vtbuf + bh * HDc * Nn;
  const float*  Bi = biasbuf + (size_t)h * Nn * Nn;
  bf16_t* P = &pshm[wave][0];

  // Q A-fragment (rows q0..q0+15, HD=32)
  v16bf qf;
  {
    const bf16_t* p = Q + (size_t)(q0 + l15) * HDc + half * 16;
    v8bf lo = *reinterpret_cast<const v8bf*>(p);
    v8bf hi = *reinterpret_cast<const v8bf*>(p + 8);
    qf = __builtin_shufflevector(lo, hi, 0,1,2,3,4,5,6,7,8,9,10,11,12,13,14,15);
  }

  v8f o0 = {}, o1 = {};
  float mrow[8], lrow[8];
#pragma unroll
  for (int j = 0; j < 8; ++j) { mrow[j] = -1e30f; lrow[j] = 0.f; }

  for (int kt = 0; kt < Nn; kt += 32) {
    // K B-fragments: col=key, contraction=d (contiguous per lane)
    v16bf kf0 = *reinterpret_cast<const v16bf*>(Kp + (size_t)(kt + l15) * HDc + half * 16);
    v16bf kf1 = *reinterpret_cast<const v16bf*>(Kp + (size_t)(kt + 16 + l15) * HDc + half * 16);
    v8f s0 = {}, s1 = {};
    s0 = WMMA_BF16(qf, kf0, s0);
    s1 = WMMA_BF16(qf, kf1, s1);

    // online softmax with fused (rel+global) bias
    int colk = kt + l15;
#pragma unroll
    for (int j = 0; j < 8; ++j) {
      int qr = q0 + half * 8 + j;
      const float* bp = Bi + (size_t)qr * Nn + colk;
      float v0 = s0[j] + bp[0];
      float v1 = s1[j] + bp[16];
      float t = fmaxf(v0, v1);
#pragma unroll
      for (int m = 8; m > 0; m >>= 1) t = fmaxf(t, __shfl_xor(t, m, 32)); // stays within 16-lane half
      float nm  = fmaxf(mrow[j], t);
      float p0  = __expf(v0 - nm);
      float p1  = __expf(v1 - nm);
      float rs  = p0 + p1;
#pragma unroll
      for (int m = 8; m > 0; m >>= 1) rs += __shfl_xor(rs, m, 32);
      float fac = __expf(mrow[j] - nm);
      lrow[j] = lrow[j] * fac + rs;
      o0[j] *= fac;  o1[j] *= fac;
      mrow[j] = nm;
      P[(half * 8 + j) * 32 + l15]      = f2bf(p0);
      P[(half * 8 + j) * 32 + 16 + l15] = f2bf(p1);
    }
    __builtin_amdgcn_wave_barrier();
    asm volatile("s_wait_dscnt 0x0" ::: "memory");

    // P A-fragment (queries x 32 keys) from LDS
    v16bf pf;
    {
      const bf16_t* p = P + (size_t)l15 * 32 + half * 16;
      v8bf lo = *reinterpret_cast<const v8bf*>(p);
      v8bf hi = *reinterpret_cast<const v8bf*>(p + 8);
      pf = __builtin_shufflevector(lo, hi, 0,1,2,3,4,5,6,7,8,9,10,11,12,13,14,15);
    }
    // V B-fragments: col=d, contraction=key (contiguous thanks to transposed V)
    v16bf vf0 = *reinterpret_cast<const v16bf*>(Vt + (size_t)l15 * Nn + kt + half * 16);
    v16bf vf1 = *reinterpret_cast<const v16bf*>(Vt + (size_t)(16 + l15) * Nn + kt + half * 16);
    o0 = WMMA_BF16(pf, vf0, o0);
    o1 = WMMA_BF16(pf, vf1, o1);
    __builtin_amdgcn_wave_barrier();
  }

  // finalize and write [b*N+q, h*32+d] as bf16 for proj GEMM
#pragma unroll
  for (int j = 0; j < 8; ++j) {
    float inv = 1.f / lrow[j];
    int qr = q0 + half * 8 + j;
    size_t base = ((size_t)b * Nn + qr) * Cc + h * HDc;
    outb[base + l15]      = f2bf(o0[j] * inv);
    outb[base + 16 + l15] = f2bf(o1[j] * inv);
  }
}

// ---------------------------------------------------------------------------
extern "C" void kernel_launch(void* const* d_in, const int* in_sizes, int n_in,
                              void* d_out, int out_size, void* d_ws, size_t ws_size,
                              hipStream_t stream) {
  (void)in_sizes; (void)n_in; (void)out_size; (void)ws_size;
  const float* x          = (const float*)d_in[0];
  const int*   rel_index  = (const int*)  d_in[3];
  const float* ln1_g      = (const float*)d_in[4];
  const float* ln1_b      = (const float*)d_in[5];
  const float* w_qkv      = (const float*)d_in[6];
  const float* w_proj     = (const float*)d_in[7];
  const float* b_proj     = (const float*)d_in[8];
  const float* rel_table  = (const float*)d_in[9];
  const float* gbias      = (const float*)d_in[10];
  const float* ln2_g      = (const float*)d_in[11];
  const float* ln2_b      = (const float*)d_in[12];
  const float* w_pw1      = (const float*)d_in[13];
  const float* b_pw1      = (const float*)d_in[14];
  const float* w_dw       = (const float*)d_in[15];
  const float* b_dw       = (const float*)d_in[16];
  const float* w_pw2      = (const float*)d_in[17];
  const float* b_pw2      = (const float*)d_in[18];
  float* out = (float*)d_out;

  const int ROWS = Bb * Nn;          // 8192
  char* p = (char*)d_ws;
  auto alloc = [&](size_t bytes) { void* r = p; p += (bytes + 255) & ~(size_t)255; return r; };
  bf16_t* lnbuf  = (bf16_t*)alloc((size_t)ROWS * Cc   * 2);   // LN2(x), later LN1(x1)
  bf16_t* act1   = (bf16_t*)alloc((size_t)ROWS * HIDc * 2);
  bf16_t* act2   = (bf16_t*)alloc((size_t)ROWS * HIDc * 2);
  float*  x1     = (float*) alloc((size_t)ROWS * Cc   * 4);
  bf16_t* qb     = (bf16_t*)alloc((size_t)Bb * Hh * Nn * HDc * 2);
  bf16_t* kb     = (bf16_t*)alloc((size_t)Bb * Hh * Nn * HDc * 2);
  bf16_t* vtb    = (bf16_t*)alloc((size_t)Bb * Hh * HDc * Nn * 2);
  bf16_t* attout = (bf16_t*)alloc((size_t)ROWS * Cc * 2);
  float*  biasb  = (float*) alloc((size_t)Hh * Nn * Nn * 4);
  bf16_t* wpw1b  = (bf16_t*)alloc((size_t)HIDc * Cc   * 2);
  bf16_t* wpw2b  = (bf16_t*)alloc((size_t)Cc   * HIDc * 2);
  bf16_t* wqkvb  = (bf16_t*)alloc((size_t)3 * Cc * Cc * 2);
  bf16_t* wprojb = (bf16_t*)alloc((size_t)Cc * Cc     * 2);

  // weight bf16 conversion + fused attention bias
  k_cvt_bf16<<<(HIDc * Cc + 255) / 256, 256, 0, stream>>>(w_pw1, wpw1b, HIDc * Cc);
  k_cvt_bf16<<<(Cc * HIDc + 255) / 256, 256, 0, stream>>>(w_pw2, wpw2b, Cc * HIDc);
  k_cvt_bf16<<<(3 * Cc * Cc + 255) / 256, 256, 0, stream>>>(w_qkv, wqkvb, 3 * Cc * Cc);
  k_cvt_bf16<<<(Cc * Cc + 255) / 256, 256, 0, stream>>>(w_proj, wprojb, Cc * Cc);
  k_bias_combine<<<(Nn * Nn + 255) / 256, 256, 0, stream>>>(rel_index, rel_table, gbias, biasb);

  // --- MLP branch: x1 = x + pw2(gelu(dw(gelu(pw1(LN2 x))))) ---
  k_ln_bf16<<<ROWS / 8, 256, 0, stream>>>(x, ln2_g, ln2_b, lnbuf);
  k_gemm_gelu_bf16<<<dim3(HIDc / 128, ROWS / 128), 256, 0, stream>>>(lnbuf, wpw1b, b_pw1, act1, Cc, HIDc);
  k_dwconv_gelu<<<(Bb * Nn * HIDc + 255) / 256, 256, 0, stream>>>(act1, w_dw, b_dw, act2);
  k_gemm_bias_res_f32<<<dim3(Cc / 128, ROWS / 128), 256, 0, stream>>>(act2, wpw2b, b_pw2, x, x1, HIDc, Cc);

  // --- attention branch: out = x1 + proj(attn(LN1 x1)) ---
  k_ln_bf16<<<ROWS / 8, 256, 0, stream>>>(x1, ln1_g, ln1_b, lnbuf);
  k_gemm_qkv<<<dim3(3 * Cc / 128, ROWS / 128), 256, 0, stream>>>(lnbuf, wqkvb, qb, kb, vtb);
  k_flash_attn<<<dim3(Nn / 64, Hh, Bb), 128, 0, stream>>>(qb, kb, vtb, biasb, attout);
  k_gemm_bias_res_f32<<<dim3(Cc / 128, ROWS / 128), 256, 0, stream>>>(attout, wprojb, b_proj, x1, out, Cc, Cc);
}
